// CPN_69879117906403
// MI455X (gfx1250) — matrix operand order repaired and checked by reading, not compiled, not run on hardware
//
#include <hip/hip_runtime.h>
#include <math.h>

// Problem constants (from reference)
constexpr int BB   = 8;      // batch
constexpr int NN   = 2048;   // points
constexpr int PP   = 8;      // patches
constexpr int SS   = 256;    // points per patch
constexpr int DD   = 1026;   // morph feature dim
constexpr int LDIM = 1024;   // latent dim
constexpr int NCC  = 2048;   // coarse points  (PP*SS)
constexpr int MM   = 4096;   // dense0 points  (NN+NCC)
constexpr float ALPHA_C = 1.5f;

typedef __attribute__((ext_vector_type(16))) __bf16 v16bf;
typedef __attribute__((ext_vector_type(8)))  float  v8f;

__device__ __forceinline__ unsigned pack2bf(float lo, float hi) {
  union { unsigned u; __bf16 b[2]; } r;
  r.b[0] = (__bf16)lo; r.b[1] = (__bf16)hi;
  return r.u;
}

__device__ __forceinline__ float act_apply(float v, int act) {
  if (act == 1) return fmaxf(v, 0.f);
  if (act == 2) return v > 0.f ? v : expm1f(v);      // ELU(alpha=1)
  return v;
}

// ---------------------------------------------------------------------------
// Weight pre-conversion: fp32 [batch,M,K] -> bf16 [batch,Mp,Kp], zero padded.
// ---------------------------------------------------------------------------
__global__ void wconv_kernel(const float* __restrict__ src, unsigned short* __restrict__ dst,
                             int M, int K, int Mp, int Kp, int batch) {
  long long i = (long long)blockIdx.x * 256 + threadIdx.x;
  if (i >= (long long)batch * Mp * Kp) return;
  int p = (int)(i / ((long long)Mp * Kp));
  int rem = (int)(i % ((long long)Mp * Kp));
  int m = rem / Kp, k = rem % Kp;
  float v = (m < M && k < K) ? src[((long long)p * M + m) * K + k] : 0.f;
  union { unsigned short s; __bf16 b; } c; c.b = (__bf16)v;
  dst[i] = c.s;
}

// ---------------------------------------------------------------------------
// Generic batched GEMM:  D[z] = act(A[z/aDiv] @ B[z]) (+= existing D)
// FULLY PADDED: A bf16 [Mp,Kp] (Mp = gridDim.y*64, Kp mult of 32, zero pad);
// B fp32 [Kp,N] rows >= Kp allocated, pad rows zero; N mult of 64.
// D fp32 [Mp,N]: ALL Mp rows written (pad rows get act(0)=0).
// bf16 WMMA 16x16x32, block tile 64x64, 4 waves, 2x2 frags, double-buffered LDS.
// A operand staged via GLOBAL_LOAD_ASYNC_TO_LDS_B128 (ASYNCcnt), B staged via
// b128 loads + v_cvt_pk_bf16_f32 + transposed packed ds stores.
// ---------------------------------------------------------------------------
__global__ __launch_bounds__(128)
void wmma_gemm_kernel(const unsigned short* __restrict__ A, const float* __restrict__ Bm,
                      float* __restrict__ Dm,
                      int N, int Kp,
                      long long sA, long long sB, long long sD,
                      int aDiv, int act, int addOut)
{
  __shared__ __align__(16) unsigned short As[2][64 * 32];   // [m][k] bf16
  __shared__ __align__(16) unsigned short Bs[2][64 * 32];   // [n][k] bf16 (transposed)

  const int z = blockIdx.z;
  A  += (long long)(z / aDiv) * sA;
  Bm += (long long)z * sB;
  Dm += (long long)z * sD;

  const int tid  = threadIdx.x;
  const int lane = tid & 31;
  const int wave = tid >> 5;
  const int wm = wave >> 1, wn = wave & 1;
  const int mTile = blockIdx.y * 64, nTile = blockIdx.x * 64;

  // A staging: 16 bf16 (32B) per thread; B staging: two k-rows x 8 cols
  const int ar = tid >> 1, ah = tid & 1;
  const int bkp = tid >> 3, bn0 = (tid & 7) * 8;
  const unsigned short* aBase = A + (long long)(mTile + ar) * Kp + ah * 16;
  const float* bBase = Bm + (long long)(2 * bkp) * N + nTile + bn0;

  v8f acc[2][2];
  #pragma unroll
  for (int i = 0; i < 2; ++i)
    #pragma unroll
    for (int j = 0; j < 2; ++j)
      #pragma unroll
      for (int r = 0; r < 8; ++r) acc[i][j][r] = 0.f;

  // Async DMA copy of the (pre-converted, padded) bf16 A tile into LDS.
  auto stageA = [&](int buf, int k0) {
    unsigned lds_off = (unsigned)(unsigned long long)&As[buf][ar * 32 + ah * 16];
    const unsigned short* gsrc = aBase + k0;
    asm volatile(
        "global_load_async_to_lds_b128 %0, %1, off\n\t"
        "global_load_async_to_lds_b128 %0, %1, off offset:16"
        :: "v"(lds_off), "v"(gsrc) : "memory");
  };

  auto stageB = [&](int buf, int k0) {
    const float* p0 = bBase + (long long)k0 * N;
    const float* p1 = p0 + N;
    float4 x0 = ((const float4*)p0)[0], y0 = ((const float4*)p0)[1];
    float4 x1 = ((const float4*)p1)[0], y1 = ((const float4*)p1)[1];
    float r0[8] = {x0.x, x0.y, x0.z, x0.w, y0.x, y0.y, y0.z, y0.w};
    float r1[8] = {x1.x, x1.y, x1.z, x1.w, y1.x, y1.y, y1.z, y1.w};
    #pragma unroll
    for (int q = 0; q < 8; ++q)
      *(unsigned*)(&Bs[buf][(bn0 + q) * 32 + 2 * bkp]) = pack2bf(r0[q], r1[q]);
  };

  auto compute = [&](int buf) {
    union FU { v16bf v; unsigned u[8]; };
    FU af[2], bfr[2];
    const int mrow = lane & 15;
    // A 16x32 bf16 layout: lanes0-15 K {0..7,16..23}, lanes16-31 K {8..15,24..31}
    const int akb = (lane < 16) ? 0 : 8;
    #pragma unroll
    for (int i = 0; i < 2; ++i) {
      int mbase = (wm * 32 + i * 16 + mrow) * 32;
      #pragma unroll
      for (int p = 0; p < 8; ++p) {
        int kk = akb + ((p < 4) ? (2 * p) : (8 + 2 * p));  // dword = 2 consecutive K
        af[i].u[p] = *(const unsigned*)(&As[buf][mbase + kk]);
      }
    }
    // B 32x16 bf16 layout: lanes0-15 K {0..7,16..23} (N=lane), lanes16-31 K {8..15,24..31}
    const int bkb = (lane < 16) ? 0 : 8;
    #pragma unroll
    for (int j = 0; j < 2; ++j) {
      int nbase = (wn * 32 + j * 16 + mrow) * 32;
      #pragma unroll
      for (int p = 0; p < 8; ++p)
        bfr[j].u[p] = *(const unsigned*)(&Bs[buf][nbase + bkb + ((p < 4) ? (2 * p) : (8 + 2 * p))]);
    }
    #pragma unroll
    for (int i = 0; i < 2; ++i)
      #pragma unroll
      for (int j = 0; j < 2; ++j)
        acc[i][j] = __builtin_amdgcn_wmma_f32_16x16x32_bf16(
            false, af[i].v, false, bfr[j].v, (short)0, acc[i][j], false, false);
  };

  stageA(0, 0);
  stageB(0, 0);
  int buf = 0;
  for (int k0 = 0; k0 < Kp; k0 += 32) {
    asm volatile("s_wait_asynccnt 0x0" ::: "memory");   // A DMA for 'buf' done
    __syncthreads();                                    // + B stores visible
    if (k0 + 32 < Kp) {                                 // overlap next stage
      stageA(buf ^ 1, k0 + 32);
      stageB(buf ^ 1, k0 + 32);
    }
    compute(buf);
    buf ^= 1;
  }

  // Epilogue (no guards: everything padded). C/D layout: VGPR r -> row r /(8+r)
  const int col0 = nTile + wn * 32 + (lane & 15);
  const int row0 = mTile + wm * 32 + ((lane >> 4) << 3);
  #pragma unroll
  for (int i = 0; i < 2; ++i)
    #pragma unroll
    for (int j = 0; j < 2; ++j)
      #pragma unroll
      for (int r = 0; r < 8; ++r) {
        float v = act_apply(acc[i][j][r], act);
        float* dp = Dm + (long long)(row0 + i * 16 + r) * N + (col0 + j * 16);
        if (addOut) v += *dp;
        *dp = v;
      }
}

// ---------------------------------------------------------------------------
// Small helper kernels
// ---------------------------------------------------------------------------
// xyz -> f0 padded [b][32][n]  (rows 6..31 zero, so stage-1 K pads cleanly)
__global__ void transpose_kernel(const float* __restrict__ xyz, float* __restrict__ f0p) {
  int i = blockIdx.x * 256 + threadIdx.x;
  if (i >= BB * 32 * NN) return;
  int b = i / (32 * NN), rem = i % (32 * NN), c = rem / NN, n = rem % NN;
  f0p[i] = (c < 6) ? xyz[((long long)b * NN + n) * 6 + c] : 0.f;
}

__global__ void minmax_kernel(const float* __restrict__ xyz, float* cmin, float* cmax) {
  int bd = blockIdx.x, b = bd / 3, d = bd % 3;
  __shared__ float smin[256], smax[256];
  float mn = 3e38f, mx = -3e38f;
  for (int n = threadIdx.x; n < NN; n += 256) {
    float v = xyz[((long long)b * NN + n) * 6 + d];
    mn = fminf(mn, v); mx = fmaxf(mx, v);
  }
  smin[threadIdx.x] = mn; smax[threadIdx.x] = mx; __syncthreads();
  for (int o = 128; o > 0; o >>= 1) {
    if (threadIdx.x < o) {
      smin[threadIdx.x] = fminf(smin[threadIdx.x], smin[threadIdx.x + o]);
      smax[threadIdx.x] = fmaxf(smax[threadIdx.x], smax[threadIdx.x + o]);
    }
    __syncthreads();
  }
  if (threadIdx.x == 0) { cmin[bd] = smin[0]; cmax[bd] = smax[0]; }
}

__global__ void prep_kernel(const float* __restrict__ xyz, const float* cmin, const float* cmax,
                            int R, int* __restrict__ flat, int* __restrict__ cidx,
                            float* __restrict__ cw) {
  int i = blockIdx.x * 256 + threadIdx.x;
  if (i >= BB * NN) return;
  int b = i / NN, n = i % NN;
  int vi[3], c0[3]; float fr[3];
  for (int d = 0; d < 3; ++d) {
    float lo = cmin[b * 3 + d], hi = cmax[b * 3 + d];
    float x = xyz[((long long)b * NN + n) * 6 + d];
    float v = (x - lo) / (hi - lo + 1e-8f);
    int t = (int)(v * (float)R);
    vi[d] = min(max(t, 0), R - 1);
    float cc = v * (float)(R - 1);
    int f0 = (int)floorf(cc); f0 = min(max(f0, 0), R - 1);
    c0[d] = f0; fr[d] = cc - (float)f0;
  }
  flat[i] = (vi[0] * R + vi[1]) * R + vi[2];
  for (int dx = 0; dx < 2; ++dx)
    for (int dy = 0; dy < 2; ++dy)
      for (int dz = 0; dz < 2; ++dz) {
        int k = dx * 4 + dy * 2 + dz;
        int ix = min(c0[0] + dx, R - 1), iy = min(c0[1] + dy, R - 1), iz = min(c0[2] + dz, R - 1);
        float w = (dx ? fr[0] : 1.f - fr[0]) * (dy ? fr[1] : 1.f - fr[1]) * (dz ? fr[2] : 1.f - fr[2]);
        cidx[i * 8 + k] = (ix * R + iy) * R + iz;
        cw[i * 8 + k] = w;
      }
}

// f rows live in a layout with row-stride Cs (>= Cin): index (b*Cs+ci)*NN+n
__global__ void scatter_f_kernel(const float* __restrict__ f, const int* __restrict__ flat,
                                 float* __restrict__ vox, int Cin, int Cs, int R3) {
  int i = blockIdx.x * 256 + threadIdx.x;
  if (i >= BB * Cin * NN) return;
  int b = i / (Cin * NN), rem = i % (Cin * NN), ci = rem / NN, n = rem % NN;
  atomicAdd(&vox[((long long)b * Cin + ci) * R3 + flat[b * NN + n]],
            f[((long long)b * Cs + ci) * NN + n]);
}

__global__ void scatter_cnt_kernel(const int* __restrict__ flat, float* __restrict__ cnt, int R3) {
  int i = blockIdx.x * 256 + threadIdx.x;
  if (i >= BB * NN) return;
  int b = i / NN;
  atomicAdd(&cnt[b * R3 + flat[i]], 1.f);
}

__global__ void voxnorm_kernel(float* __restrict__ vox, const float* __restrict__ cnt,
                               int Cin, int R3) {
  int i = blockIdx.x * 256 + threadIdx.x;
  if (i >= BB * Cin * R3) return;
  int b = i / (Cin * R3), v = i % R3;
  vox[i] /= fmaxf(cnt[b * R3 + v], 1.f);
}

// im2col (per batch) with K padded to Kp27: rows >= Cin*27 are zero
__global__ void im2col_kernel(const float* __restrict__ voxB, float* __restrict__ col,
                              int Cin, int R, int R3, int Kp27) {
  int i = blockIdx.x * 256 + threadIdx.x;
  if (i >= Kp27 * R3) return;
  int kk = i / R3, v = i % R3;
  if (kk >= Cin * 27) { col[i] = 0.f; return; }
  int ci = kk / 27, t = kk % 27;
  int a = t / 9, bq = (t % 9) / 3, cq = t % 3;
  int x = v / (R * R), y = (v / R) % R, zq = v % R;
  int xx = x + a - 1, yy = y + bq - 1, zz = zq + cq - 1;
  float val = 0.f;
  if (xx >= 0 && xx < R && yy >= 0 && yy < R && zz >= 0 && zz < R)
    val = voxB[(long long)ci * R3 + (xx * R + yy) * R + zz];
  col[i] = val;
}

// Trilinear devoxelize: fout[b,co,n] = sum_k cw * voxf[b,co,cidx]
__global__ void devox_kernel(const float* __restrict__ voxf, const int* __restrict__ cidx,
                             const float* __restrict__ cw, float* __restrict__ fout,
                             int Cout, int R3) {
  int b = blockIdx.x / Cout, co = blockIdx.x % Cout;
  const float* vslice = voxf + ((long long)b * Cout + co) * R3;
  for (int n = threadIdx.x; n < NN; n += 256) {
    const int*   ip = cidx + ((long long)b * NN + n) * 8;
    const float* wp = cw   + ((long long)b * NN + n) * 8;
    float acc = 0.f;
    #pragma unroll
    for (int k = 0; k < 8; ++k) acc += wp[k] * vslice[ip[k]];
    fout[((long long)b * Cout + co) * NN + n] = acc;
  }
}

__global__ void maxpool_kernel(const float* __restrict__ f, float* __restrict__ featRaw) {
  int bc = blockIdx.x;  // b*1024+c
  __shared__ float red[256];
  const float* p = f + (long long)bc * NN;
  float mx = -3e38f;
  for (int n = threadIdx.x; n < NN; n += 256) mx = fmaxf(mx, p[n]);
  red[threadIdx.x] = mx; __syncthreads();
  for (int o = 128; o > 0; o >>= 1) {
    if (threadIdx.x < o) red[threadIdx.x] = fmaxf(red[threadIdx.x], red[threadIdx.x + o]);
    __syncthreads();
  }
  if (threadIdx.x == 0) featRaw[bc] = red[0];
}

__global__ void lin_kernel(const float* __restrict__ featRaw, const float* __restrict__ w,
                           const float* __restrict__ bias, float* __restrict__ feat) {
  int i = blockIdx.x * 256 + threadIdx.x;
  if (i >= BB * LDIM) return;
  int b = i / LDIM, o = i % LDIM;
  const float* fr = featRaw + b * LDIM;
  const float* wr = w + (long long)o * LDIM;
  float s = bias[o];
  for (int c = 0; c < LDIM; ++c) s += fr[c] * wr[c];
  feat[i] = s > 0.f ? s : expm1f(s);
}

// y0 padded to 1056 rows per (p,b): rows [0,2) rand, [2,1026) feat, rest zero
constexpr int DP0 = 1056;
__global__ void build_y0_kernel(const float* __restrict__ rg, const float* __restrict__ feat,
                                float* __restrict__ y0) {
  long long i = (long long)blockIdx.x * 256 + threadIdx.x;
  if (i >= (long long)PP * BB * DP0 * SS) return;
  int p = (int)(i / ((long long)BB * DP0 * SS));
  int r1 = (int)(i % ((long long)BB * DP0 * SS));
  int b = r1 / (DP0 * SS), r2 = r1 % (DP0 * SS);
  int c = r2 / SS, s = r2 % SS;
  float v;
  if (c < 2)        v = rg[(((long long)p * BB + b) * 2 + c) * SS + s];
  else if (c < DD)  v = feat[b * LDIM + (c - 2)];
  else              v = 0.f;
  y0[i] = v;
}

__global__ void write_coarse_kernel(const float* __restrict__ outs, float* __restrict__ coarse) {
  int i = blockIdx.x * 256 + threadIdx.x;
  if (i >= BB * NCC * 6) return;
  int b = i / (NCC * 6), rem = i % (NCC * 6), n = rem / 6, c = rem % 6;
  int p = n / SS, s = n % SS;
  coarse[i] = outs[(((long long)p * BB + b) * 64 + c) * SS + s];  // outs padded to 64 rows
}

__global__ void nn_kernel(const float* __restrict__ coarse, float* __restrict__ nnb,
                          float* __restrict__ msum) {
  int bp = blockIdx.x;        // b*P + p
  int b = bp >> 3, s = threadIdx.x;
  __shared__ float sx[SS], sy[SS], sz[SS], red[SS];
  const float* c = coarse + ((long long)b * NCC + (bp & 7) * SS) * 6;
  sx[s] = c[s * 6 + 0]; sy[s] = c[s * 6 + 1]; sz[s] = c[s * 6 + 2];
  __syncthreads();
  float x = sx[s], y = sy[s], z = sz[s], best = 3e38f;
  for (int t = 0; t < SS; ++t) {
    if (t == s) continue;
    float dx = x - sx[t], dy = y - sy[t], dz = z - sz[t];
    best = fminf(best, dx * dx + dy * dy + dz * dz);
  }
  float nnv = sqrtf(best);
  nnb[bp * SS + s] = nnv;
  red[s] = nnv; __syncthreads();
  for (int o = 128; o > 0; o >>= 1) {
    if (s < o) red[s] += red[s + o];
    __syncthreads();
  }
  if (s == 0) atomicAdd(&msum[b], red[0]);
}

__global__ void loss_kernel(const float* __restrict__ nnb, const float* __restrict__ msum,
                            float* __restrict__ lossAcc) {
  int bp = blockIdx.x, b = bp >> 3, s = threadIdx.x;
  __shared__ float red[SS];
  float mean = msum[b] / (float)(PP * SS);
  float v = nnb[bp * SS + s];
  red[s] = (v > ALPHA_C * mean) ? v : 0.f;
  __syncthreads();
  for (int o = 128; o > 0; o >>= 1) {
    if (s < o) red[s] += red[s + o];
    __syncthreads();
  }
  if (s == 0) atomicAdd(lossAcc, red[0]);
}

__global__ void finalize_loss_kernel(const float* lossAcc, float* out) {
  *out = lossAcc[0] / (float)(BB * PP * SS);
}

__global__ void build_dense0_kernel(const float* __restrict__ xyz, const float* __restrict__ coarse,
                                    float* __restrict__ dense0) {
  int i = blockIdx.x * 256 + threadIdx.x;
  if (i >= BB * MM * 6) return;
  int b = i / (MM * 6), rem = i % (MM * 6), r = rem / 6, c = rem % 6;
  dense0[i] = (r < NN) ? xyz[((long long)b * NN + r) * 6 + c]
                       : coarse[((long long)b * NCC + (r - NN)) * 6 + c];
}

// Greedy density-based selection: one block per batch, sequential M steps.
__global__ __launch_bounds__(1024)
void mds_kernel(const float* __restrict__ dense0, const float* __restrict__ msum,
                int* __restrict__ idxs) {
  const int b = blockIdx.x, tid = threadIdx.x;
  float sigma = fmaxf(msum[b] / (float)(PP * SS), 1e-6f);
  float inv2 = 1.f / (2.f * sigma * sigma);
  const float* base = dense0 + (long long)b * MM * 6;
  float px[4], py[4], pz[4], dens[4]; int sel[4];
  const int j0 = tid * 4;
  #pragma unroll
  for (int k = 0; k < 4; ++k) {
    const float* q = base + (long long)(j0 + k) * 6;
    px[k] = q[0]; py[k] = q[1]; pz[k] = q[2];
    dens[k] = 0.f; sel[k] = 0;
  }
  __shared__ float rv[32]; __shared__ int ri[32]; __shared__ float pick[3];
  for (int iter = 0; iter < MM; ++iter) {
    float bv = 3e38f; int bi = 0x7fffffff;
    #pragma unroll
    for (int k = 0; k < 4; ++k) {
      float v = dens[k] + (sel[k] ? 1e10f : 0.f);
      int jj = j0 + k;
      if (v < bv || (v == bv && jj < bi)) { bv = v; bi = jj; }
    }
    for (int off = 16; off > 0; off >>= 1) {     // wave32 reduce, first-index ties
      float ov = __shfl_down(bv, off);
      int   oi = __shfl_down(bi, off);
      if (ov < bv || (ov == bv && oi < bi)) { bv = ov; bi = oi; }
    }
    if ((tid & 31) == 0) { rv[tid >> 5] = bv; ri[tid >> 5] = bi; }
    __syncthreads();
    if (tid < 32) {
      bv = rv[tid]; bi = ri[tid];
      for (int off = 16; off > 0; off >>= 1) {
        float ov = __shfl_down(bv, off);
        int   oi = __shfl_down(bi, off);
        if (ov < bv || (ov == bv && oi < bi)) { bv = ov; bi = oi; }
      }
      if (tid == 0) { rv[0] = bv; ri[0] = bi; }
    }
    __syncthreads();
    int widx = ri[0];
    if (widx >= j0 && widx < j0 + 4) {
      int k = widx - j0;
      pick[0] = px[k]; pick[1] = py[k]; pick[2] = pz[k];
      sel[k] = 1;
      idxs[b * MM + iter] = widx;
    }
    __syncthreads();
    float qx = pick[0], qy = pick[1], qz = pick[2];
    #pragma unroll
    for (int k = 0; k < 4; ++k) {
      float dx = px[k] - qx, dy = py[k] - qy, dz = pz[k] - qz;
      dens[k] += __expf(-(dx * dx + dy * dy + dz * dz) * inv2);
    }
    __syncthreads();
  }
}

__global__ void gather_dense_kernel(const float* __restrict__ dense0, const int* __restrict__ idxs,
                                    float* __restrict__ dense) {
  int i = blockIdx.x * 256 + threadIdx.x;
  if (i >= BB * MM * 6) return;
  int b = i / (MM * 6), rem = i % (MM * 6), r = rem / 6, c = rem % 6;
  dense[i] = dense0[((long long)b * MM + idxs[b * MM + r]) * 6 + c];
}

// ---------------------------------------------------------------------------
extern "C" void kernel_launch(void* const* d_in, const int* in_sizes, int n_in,
                              void* d_out, int out_size, void* d_ws, size_t ws_size,
                              hipStream_t stream)
{
  (void)in_sizes; (void)n_in; (void)out_size; (void)ws_size;
  const float* xyz  = (const float*)d_in[0];
  const float* rg   = (const float*)d_in[1];
  const float* wp[4] = {(const float*)d_in[2], (const float*)d_in[4],
                        (const float*)d_in[6], (const float*)d_in[8]};
  const float* wv[4] = {(const float*)d_in[3], (const float*)d_in[5],
                        (const float*)d_in[7], (const float*)d_in[9]};
  const float* lin_w = (const float*)d_in[10];
  const float* lin_b = (const float*)d_in[11];
  const float* dw[4] = {(const float*)d_in[12], (const float*)d_in[13],
                        (const float*)d_in[14], (const float*)d_in[15]};

  float* coarse  = (float*)d_out;                      // [B,NC,6]
  float* dense   = coarse + BB * NCC * 6;              // [B,M,6]
  float* lossOut = dense + BB * MM * 6;                // scalar

  char* cur = (char*)d_ws;
  auto alloc = [&](size_t bytes) -> void* {
    void* r = (void*)cur;
    cur += (bytes + 255) & ~(size_t)255;
    return r;
  };
  // fA/fB hold the biggest of: f1..f4 (<=16.78M floats), y0 (64*1056*256),
  // y1 (64*1088*256 = 17,825,792 floats)
  float* fA     = (float*)alloc((size_t)17825792 * 4);
  float* fB     = (float*)alloc((size_t)17825792 * 4);
  float* f0p    = (float*)alloc((size_t)BB * 32 * NN * 4);   // padded stage-1 input
  float* y2buf  = (float*)alloc((size_t)64 * 576 * 256 * 4); // 9,437,184 floats
  float* colb   = (float*)alloc((size_t)7077888 * 4);        // im2col / y3 (64*256*256)
  float* vox    = (float*)alloc((size_t)2097152 * 4);
  float* cnt    = (float*)alloc((size_t)32768 * 4);
  float* voxf   = (float*)alloc((size_t)4194304 * 4);
  float* cmin   = (float*)alloc(24 * 4);
  float* cmax   = (float*)alloc(24 * 4);
  int*   flat   = (int*)alloc((size_t)16384 * 4);
  int*   cidx   = (int*)alloc((size_t)131072 * 4);
  float* cw     = (float*)alloc((size_t)131072 * 4);
  float* featRaw= (float*)alloc(8192 * 4);
  float* feat   = (float*)alloc(8192 * 4);
  float* outs   = (float*)alloc((size_t)64 * 64 * 256 * 4);  // padded to 64 rows
  float* nnb    = (float*)alloc((size_t)16384 * 4);
  float* msum   = (float*)alloc(8 * 4);
  float* lossAcc= (float*)alloc(256);
  float* dense0 = (float*)alloc((size_t)196608 * 4);
  int*   idxs   = (int*)alloc((size_t)32768 * 4);

  // bf16 weight buffers (padded Mp=ceil64, Kp=ceil32)
  const int Ci[4] = {6, 64, 128, 512};
  const int Co[4] = {64, 128, 512, 1024};
  const int Rs[4] = {16, 16, 8, 4};
  const int wvK[4]  = {162, 1728, 3456, 13824};
  const int wvKp[4] = {192, 1728, 3456, 13824};
  const int wpKp[4] = {32, 64, 128, 512};
  unsigned short* wvb[4]; unsigned short* wpb[4];
  for (int s = 0; s < 4; ++s) {
    wvb[s] = (unsigned short*)alloc((size_t)Co[s] * wvKp[s] * 2);
    wpb[s] = (unsigned short*)alloc((size_t)Co[s] * wpKp[s] * 2);
  }
  const int dwM[4]  = {1026, 513, 256, 6};
  const int dwK[4]  = {1026, 1026, 513, 256};
  const int dwMp[4] = {1088, 576, 256, 64};
  const int dwKp[4] = {1056, 1056, 544, 256};
  unsigned short* dwb[4];
  for (int l = 0; l < 4; ++l)
    dwb[l] = (unsigned short*)alloc((size_t)PP * dwMp[l] * dwKp[l] * 2);

  auto cdiv = [](long long a, long long b) { return (int)((a + b - 1) / b); };

  // --- pre-convert all weights to padded bf16 (bandwidth-trivial) ---
  for (int s = 0; s < 4; ++s) {
    wconv_kernel<<<cdiv((long long)Co[s] * wvKp[s], 256), 256, 0, stream>>>(
        wv[s], wvb[s], Co[s], wvK[s], Co[s], wvKp[s], 1);
    wconv_kernel<<<cdiv((long long)Co[s] * wpKp[s], 256), 256, 0, stream>>>(
        wp[s], wpb[s], Co[s], Ci[s], Co[s], wpKp[s], 1);
  }
  for (int l = 0; l < 4; ++l)
    wconv_kernel<<<cdiv((long long)PP * dwMp[l] * dwKp[l], 256), 256, 0, stream>>>(
        dw[l], dwb[l], dwM[l], dwK[l], dwMp[l], dwKp[l], PP);

  transpose_kernel<<<cdiv(BB * 32 * NN, 256), 256, 0, stream>>>(xyz, f0p);
  minmax_kernel<<<24, 256, 0, stream>>>(xyz, cmin, cmax);

  float* fin = f0p; float* fout = fA;
  for (int s = 0; s < 4; ++s) {
    const int R = Rs[s], R3 = R * R * R;
    const int Cs = (s == 0) ? 32 : Ci[s];   // row stride of fin
    prep_kernel<<<cdiv(BB * NN, 256), 256, 0, stream>>>(xyz, cmin, cmax, R, flat, cidx, cw);
    hipMemsetAsync(vox, 0, (size_t)BB * Ci[s] * R3 * 4, stream);
    hipMemsetAsync(cnt, 0, (size_t)BB * R3 * 4, stream);
    scatter_f_kernel<<<cdiv(BB * Ci[s] * NN, 256), 256, 0, stream>>>(fin, flat, vox,
                                                                     Ci[s], Cs, R3);
    scatter_cnt_kernel<<<cdiv(BB * NN, 256), 256, 0, stream>>>(flat, cnt, R3);
    voxnorm_kernel<<<cdiv(BB * Ci[s] * R3, 256), 256, 0, stream>>>(vox, cnt, Ci[s], R3);
    for (int b = 0; b < BB; ++b) {  // per-batch implicit-GEMM conv (stream-serialized)
      im2col_kernel<<<cdiv((long long)wvKp[s] * R3, 256), 256, 0, stream>>>(
          vox + (size_t)b * Ci[s] * R3, colb, Ci[s], R, R3, wvKp[s]);
      dim3 g(cdiv(R3, 64), cdiv(Co[s], 64), 1);
      wmma_gemm_kernel<<<g, 128, 0, stream>>>(wvb[s], colb, voxf + (size_t)b * Co[s] * R3,
          R3, wvKp[s], 0, 0, 0, 1, /*relu*/1, 0);
    }
    devox_kernel<<<BB * Co[s], 256, 0, stream>>>(voxf, cidx, cw, fout, Co[s], R3);
    {   // point branch: fout += relu(wp @ fin), batched over b  (B rows padded)
      dim3 g(cdiv(NN, 64), cdiv(Co[s], 64), BB);
      wmma_gemm_kernel<<<g, 128, 0, stream>>>(wpb[s], fin, fout,
          NN, wpKp[s], 0, (long long)Cs * NN, (long long)Co[s] * NN, 1, 1, 1);
    }
    fin = fout;
    fout = (fin == fA) ? fB : fA;
  }
  // fin holds f4 [B,1024,N] (in fB after 4 stages: f0p->fA->fB->fA->fB)
  maxpool_kernel<<<BB * LDIM, 256, 0, stream>>>(fin, featRaw);
  lin_kernel<<<cdiv(BB * LDIM, 256), 256, 0, stream>>>(featRaw, lin_w, lin_b, feat);
  float* y0 = fout;                    // the free ping-pong buffer
  float* y1 = fin;                     // f4 buffer, reusable after maxpool
  build_y0_kernel<<<cdiv((long long)PP * BB * DP0 * SS, 256), 256, 0, stream>>>(rg, feat, y0);

  { dim3 g(cdiv(SS, 64), 17, PP * BB);   // dw1: [1088p x 1056p] @ y0[1056 x 256]
    wmma_gemm_kernel<<<g, 128, 0, stream>>>(dwb[0], y0, y1, SS, dwKp[0],
        (long long)dwMp[0] * dwKp[0], (long long)DP0 * SS, (long long)1088 * SS, BB, 2, 0); }
  { dim3 g(cdiv(SS, 64), 9, PP * BB);    // dw2: [576p x 1056p] @ y1[1056 x 256]
    wmma_gemm_kernel<<<g, 128, 0, stream>>>(dwb[1], y1, y2buf, SS, dwKp[1],
        (long long)dwMp[1] * dwKp[1], (long long)1088 * SS, (long long)576 * SS, BB, 2, 0); }
  { dim3 g(cdiv(SS, 64), 4, PP * BB);    // dw3: [256 x 544p] @ y2[544 x 256]
    wmma_gemm_kernel<<<g, 128, 0, stream>>>(dwb[2], y2buf, colb, SS, dwKp[2],
        (long long)dwMp[2] * dwKp[2], (long long)576 * SS, (long long)256 * SS, BB, 2, 0); }
  { dim3 g(cdiv(SS, 64), 1, PP * BB);    // dw4: [64p x 256] @ y3[256 x 256]
    wmma_gemm_kernel<<<g, 128, 0, stream>>>(dwb[3], colb, outs, SS, dwKp[3],
        (long long)dwMp[3] * dwKp[3], (long long)256 * SS, (long long)64 * SS, BB, 2, 0); }

  write_coarse_kernel<<<cdiv(BB * NCC * 6, 256), 256, 0, stream>>>(outs, coarse);
  hipMemsetAsync(msum, 0, 8 * 4, stream);
  hipMemsetAsync(lossAcc, 0, 4, stream);
  nn_kernel<<<BB * PP, 256, 0, stream>>>(coarse, nnb, msum);
  loss_kernel<<<BB * PP, 256, 0, stream>>>(nnb, msum, lossAcc);
  finalize_loss_kernel<<<1, 1, 0, stream>>>(lossAcc, lossOut);
  build_dense0_kernel<<<cdiv(BB * MM * 6, 256), 256, 0, stream>>>(xyz, coarse, dense0);
  mds_kernel<<<BB, 1024, 0, stream>>>(dense0, msum, idxs);
  gather_dense_kernel<<<cdiv(BB * MM * 6, 256), 256, 0, stream>>>(dense0, idxs, dense);
}
